// GraphAggregator_34857954574811
// MI455X (gfx1250) — matrix operand order repaired and checked by reading, not compiled, not run on hardware
//
#include <hip/hip_runtime.h>
#include <hip/hip_bf16.h>

// ---------- types ----------
typedef __attribute__((ext_vector_type(16))) __bf16          v16bf;
typedef __attribute__((ext_vector_type(16))) unsigned short  v16u;
typedef __attribute__((ext_vector_type(8)))  float           v8f;

#define C_DIM 256
#define G_DIM 512
#define WGT_ELEMS (C_DIM * C_DIM)      // 65536 bf16 per weight matrix

__device__ __forceinline__ unsigned short f2bf(float f) {
    unsigned int u = __float_as_uint(f);
    u += 0x7fffu + ((u >> 16) & 1u);   // round-to-nearest-even
    return (unsigned short)(u >> 16);
}

__device__ __forceinline__ v8f wmma_bf16(v16bf a, v16bf b, v8f c) {
    // D = A(16x32 bf16) x B(32x16 bf16) + C(16x16 f32)
    return __builtin_amdgcn_wmma_f32_16x16x32_bf16(
        /*neg_a=*/false, a, /*neg_b=*/false, b,
        /*c_mod=*/(short)0, c, /*reuse_a=*/false, /*reuse_b=*/false);
}

// A-fragment (16x32, 16-bit): lane&15 = M row; lane>=16 uses K base +8;
// element e<8 -> K = k0+kb+e ; e>=8 -> K = k0+kb+16+(e-8)
__device__ __forceinline__ v16bf make_afrag(const float* __restrict__ rowp,
                                            int k0, float scale) {
    v16u u;
#pragma unroll
    for (int i = 0; i < 8; ++i)  u[i]     = f2bf(rowp[k0 + i] * scale);
#pragma unroll
    for (int i = 0; i < 8; ++i)  u[8 + i] = f2bf(rowp[k0 + 16 + i] * scale);
    return __builtin_bit_cast(v16bf, u);
}

// ---------------------------------------------------------------------------
// Kernel 0: zero accumulators + repack weights into bf16 B-fragment layout.
// B-fragment (32x16, 16-bit): lane<16 holds K=0..7,16..23, col=lane;
//                             lane>=16 holds K=8..15,24..31, col=lane-16.
// Storage: [(ntile*8 + ktile)*32 + lane]*16 + e  (ushort elements)
// ---------------------------------------------------------------------------
__global__ void prep_kernel(const float* __restrict__ Wlin,
                            const float* __restrict__ Wgate,
                            const float* __restrict__ Wfin,
                            unsigned short* __restrict__ wsB,
                            float* __restrict__ sums,
                            float* __restrict__ counts) {
    int tid = blockIdx.x * blockDim.x + threadIdx.x;
    if (tid < G_DIM * C_DIM) sums[tid] = 0.0f;
    if (tid < G_DIM)         counts[tid] = 0.0f;
    if (tid < 3 * WGT_ELEMS) {
        int w = tid >> 16;                   // 0=lin 1=gate 2=fin
        int r = tid & (WGT_ELEMS - 1);
        const float* W = (w == 0) ? Wlin : (w == 1) ? Wgate : Wfin;
        int e    = r & 15;
        int lane = (r >> 4) & 31;
        int blk  = r >> 9;                   // ntile*8 + ktile
        int kt   = blk & 7;
        int nt   = blk >> 3;
        int K    = kt * 32 + ((lane & 16) ? 8 : 0) + (e & 7) + ((e & 8) ? 16 : 0);
        int Ncol = nt * 16 + (lane & 15);
        wsB[(size_t)w * WGT_ELEMS + r] = f2bf(W[K * C_DIM + Ncol]);
    }
}

// ---------------------------------------------------------------------------
// Kernel 1: per-graph node counts via LDS histogram (batch sorted, 512 bins)
// ---------------------------------------------------------------------------
__global__ void counts_kernel(const int* __restrict__ batch,
                              float* __restrict__ counts, int Ntot) {
    __shared__ float h[G_DIM];
    int tid = threadIdx.x;
    h[tid] = 0.0f; h[tid + 256] = 0.0f;
    __syncthreads();
    int start = blockIdx.x * 4096;
    for (int i = start + tid; i < start + 4096 && i < Ntot; i += 256)
        atomicAdd(&h[batch[i]], 1.0f);
    __syncthreads();
    if (h[tid]       != 0.0f) atomicAdd(&counts[tid],       h[tid]);
    if (h[tid + 256] != 0.0f) atomicAdd(&counts[tid + 256], h[tid + 256]);
}

// ---------------------------------------------------------------------------
// Kernel 2: fused  gates = softmax(xW_g + b_g), g = (xW_l + b_l)*gates,
//           sorted run-length segment-sum into sums[512][256].
// 128 threads = 4 waves; each wave owns 16 rows; block owns 64 rows.
// Gates kept as unnormalized exp in LDS; 1/sum folded in via shfl broadcast.
// ---------------------------------------------------------------------------
__global__ void __launch_bounds__(128)
main_kernel(const float* __restrict__ x, const int* __restrict__ batch,
            const unsigned short* __restrict__ wB,   // [lin | gate] bf16 frags
            const float* __restrict__ bLin, const float* __restrict__ bGate,
            float* __restrict__ sums, int Ntot) {
    __shared__ float sG[64 * C_DIM];     // 64 KB: exp(gates) -> g values
    __shared__ int   sBatch[64];

    const int tid  = threadIdx.x;
    const int lane = tid & 31;
    const int wv   = tid >> 5;
    const int rb0  = blockIdx.x * 64;

    if (tid < 64) sBatch[tid] = batch[min(rb0 + tid, Ntot - 1)];

    const int myRow  = min(rb0 + wv * 16 + (lane & 15), Ntot - 1);
    const int kbaseA = (lane & 16) ? 8 : 0;
    const int h      = lane >> 4;
    const float* xrow = x + (size_t)myRow * C_DIM;

    // preload the full 16x256 A tile as 8 bf16 fragments (stays in VGPRs)
    v16bf afrag[8];
#pragma unroll
    for (int kk = 0; kk < 8; ++kk)
        afrag[kk] = make_afrag(xrow, kk * 32 + kbaseA, 1.0f);

    float* gReg = sG + wv * 16 * C_DIM;

    // ---- gates = x @ W_gate + b_gate -> LDS ----
    const unsigned short* wG = wB + WGT_ELEMS;   // gate weight fragments
    for (int nt = 0; nt < 16; ++nt) {
        v8f acc = {};
#pragma unroll
        for (int kk = 0; kk < 8; ++kk) {
            const v16bf* bp =
                (const v16bf*)(wG + (((nt * 8 + kk) * 32 + lane) << 4));
            acc = wmma_bf16(afrag[kk], *bp, acc);
        }
        int nn = nt * 16 + (lane & 15);
        float bias = bGate[nn];
#pragma unroll
        for (int r = 0; r < 8; ++r)
            gReg[(r + 8 * h) * C_DIM + nn] = acc[r] + bias;
    }

    // ---- wave-local softmax (vectorized; 2 lanes per row, float4 LDS IO) ----
    // lane (l&15) ends up holding invSum for row (l&15); normalization is
    // applied later in the states pass (gates stay as unnormalized exp).
    float invSum;
    {
        float4* rowp = (float4*)(gReg + (lane & 15) * C_DIM + h * 128);
        float mx = -3.0e38f;
        for (int j = 0; j < 32; ++j) {
            float4 v = rowp[j];
            mx = fmaxf(mx, fmaxf(fmaxf(v.x, v.y), fmaxf(v.z, v.w)));
        }
        mx = fmaxf(mx, __shfl_xor(mx, 16));
        float s = 0.0f;
        for (int j = 0; j < 32; ++j) {
            float4 v = rowp[j];
            v.x = __expf(v.x - mx); v.y = __expf(v.y - mx);
            v.z = __expf(v.z - mx); v.w = __expf(v.w - mx);
            rowp[j] = v;
            s += v.x + v.y + v.z + v.w;
        }
        s += __shfl_xor(s, 16);
        invSum = 1.0f / s;
    }

    // ---- states = x@W_lin + b_lin ; g = states * exp * invSum (in LDS) ----
    for (int nt = 0; nt < 16; ++nt) {
        v8f acc = {};
#pragma unroll
        for (int kk = 0; kk < 8; ++kk) {
            const v16bf* bp =
                (const v16bf*)(wB + (((nt * 8 + kk) * 32 + lane) << 4));
            acc = wmma_bf16(afrag[kk], *bp, acc);
        }
        int nn = nt * 16 + (lane & 15);
        float bias = bLin[nn];
#pragma unroll
        for (int r = 0; r < 8; ++r) {
            int M = r + 8 * h;                       // row within wave tile
            float inv = __shfl(invSum, M);           // row M's 1/sum
            int off = M * C_DIM + nn;
            gReg[off] = (acc[r] + bias) * gReg[off] * inv;
        }
    }

    __syncthreads();

    // ---- sorted run-length segment-sum ----
    // 64 float4 column groups x 2 row halves; rows sorted within each half.
    {
        const int grp  = tid & 63;        // column group (4 floats)
        const int half = tid >> 6;        // 0: rows 0..31, 1: rows 32..63
        const int c4   = grp * 4;
        const int j0   = half * 32;
        int cur = sBatch[j0];
        float4 acc = make_float4(0.f, 0.f, 0.f, 0.f);
        for (int j = j0; j < j0 + 32; ++j) {
            int gid = sBatch[j];
            float4 v = *(const float4*)(sG + j * C_DIM + c4);
            if (gid != cur) {
                float* dst = sums + cur * C_DIM + c4;
                atomicAdd(dst + 0, acc.x); atomicAdd(dst + 1, acc.y);
                atomicAdd(dst + 2, acc.z); atomicAdd(dst + 3, acc.w);
                cur = gid;
                acc = make_float4(0.f, 0.f, 0.f, 0.f);
            }
            acc.x += v.x; acc.y += v.y; acc.z += v.z; acc.w += v.w;
        }
        float* dst = sums + cur * C_DIM + c4;
        atomicAdd(dst + 0, acc.x); atomicAdd(dst + 1, acc.y);
        atomicAdd(dst + 2, acc.z); atomicAdd(dst + 3, acc.w);
    }
}

// ---------------------------------------------------------------------------
// Kernel 3: out = (sums / max(counts,1)) @ W_fin + b_fin   (512x256)
// 32 blocks x 4 waves; block = one 16-row tile; wave = 4 column tiles.
// ---------------------------------------------------------------------------
__global__ void __launch_bounds__(128)
finalize_kernel(const float* __restrict__ sums, const float* __restrict__ counts,
                const unsigned short* __restrict__ wFin,
                const float* __restrict__ bFin, float* __restrict__ out) {
    const int tid  = threadIdx.x;
    const int lane = tid & 31;
    const int wv   = tid >> 5;
    const int rowBase = blockIdx.x * 16;
    const int myRow   = rowBase + (lane & 15);
    const int kbaseA  = (lane & 16) ? 8 : 0;
    const int h       = lane >> 4;

    const float inv = 1.0f / fmaxf(counts[myRow], 1.0f);
    const float* srow = sums + (size_t)myRow * C_DIM;

    v16bf afrag[8];
#pragma unroll
    for (int kk = 0; kk < 8; ++kk)
        afrag[kk] = make_afrag(srow, kk * 32 + kbaseA, inv);

#pragma unroll
    for (int t = 0; t < 4; ++t) {
        int nt = wv * 4 + t;
        v8f acc = {};
#pragma unroll
        for (int kk = 0; kk < 8; ++kk) {
            const v16bf* bp =
                (const v16bf*)(wFin + (((nt * 8 + kk) * 32 + lane) << 4));
            acc = wmma_bf16(afrag[kk], *bp, acc);
        }
        int nn = nt * 16 + (lane & 15);
        float bias = bFin[nn];
#pragma unroll
        for (int r = 0; r < 8; ++r)
            out[(rowBase + r + 8 * h) * C_DIM + nn] = acc[r] + bias;
    }
}

// ---------------------------------------------------------------------------
extern "C" void kernel_launch(void* const* d_in, const int* in_sizes, int n_in,
                              void* d_out, int out_size, void* d_ws, size_t ws_size,
                              hipStream_t stream) {
    const float* x     = (const float*)d_in[0];
    // d_in[1] = edge_index (unused by the reference)
    const int*   batch = (const int*)d_in[2];
    const float* Wlin  = (const float*)d_in[3];
    const float* bLin  = (const float*)d_in[4];
    const float* Wgate = (const float*)d_in[5];
    const float* bGate = (const float*)d_in[6];
    const float* Wfin  = (const float*)d_in[7];
    const float* bFin  = (const float*)d_in[8];
    float* out = (float*)d_out;

    const int Ntot = in_sizes[0] / C_DIM;         // 400000

    // workspace layout
    unsigned short* wsB = (unsigned short*)d_ws;          // 3*65536 bf16
    float* sums   = (float*)((char*)d_ws + 3 * WGT_ELEMS * sizeof(unsigned short));
    float* counts = sums + G_DIM * C_DIM;

    prep_kernel<<<768, 256, 0, stream>>>(Wlin, Wgate, Wfin, wsB, sums, counts);
    counts_kernel<<<(Ntot + 4095) / 4096, 256, 0, stream>>>(batch, counts, Ntot);
    main_kernel<<<Ntot / 64, 128, 0, stream>>>(x, batch, wsB, bLin, bGate, sums, Ntot);
    finalize_kernel<<<G_DIM / 16, 128, 0, stream>>>(sums, counts,
                                                    wsB + 2 * WGT_ELEMS, bFin, out);
}
